// SimpleAttention_2190433321490
// MI455X (gfx1250) — matrix operand order repaired and checked by reading, not compiled, not run on hardware
//
#include <hip/hip_runtime.h>

typedef _Float16 f16;
typedef __attribute__((ext_vector_type(16))) _Float16 v16h;
typedef __attribute__((ext_vector_type(8)))  _Float16 v8h;
typedef __attribute__((ext_vector_type(8)))  float    v8f;
typedef __attribute__((ext_vector_type(4)))  float    v4f;
typedef __attribute__((ext_vector_type(4)))  unsigned int u32x4;
typedef __attribute__((ext_vector_type(8)))  unsigned int u32x8;

#define BATCH 4
#define SEQ   2048
#define DIM   1024

// ---------------------------------------------------------------------------
// Fragment loaders (wave32 WMMA 16x16x32 f16; contiguous k order used here, same for A and B)
//   A (16x32, MxK): lane r=lane&15 -> row M=r; half h=lane>>4:
//     elems j: k = k0 + h*16 + j
//   B (32x16, KxN): lane r -> column N=r; elems j: k = k0 + h*16 + j
// ---------------------------------------------------------------------------
static __device__ __forceinline__ v16h load_a_f16(const f16* __restrict__ A, int lda,
                                                  int m0, int k0, int r, int h) {
  const f16* base = A + (size_t)(m0 + r) * lda + k0 + h * 16;
  v8h lo = *(const v8h*)(base);
  v8h hi = *(const v8h*)(base + 8);
  v16h a;
#pragma unroll
  for (int j = 0; j < 8; ++j) { a[j] = lo[j]; a[j + 8] = hi[j]; }
  return a;
}

static __device__ __forceinline__ v16h load_b_rowk_f16(const f16* __restrict__ Bm, int ldb,
                                                       int n0, int k0, int r, int h) {
  // B column n = row n of Bm (row-major [n][k]) -> contiguous along k
  const f16* base = Bm + (size_t)(n0 + r) * ldb + k0 + h * 16;
  v8h lo = *(const v8h*)(base);
  v8h hi = *(const v8h*)(base + 8);
  v16h b;
#pragma unroll
  for (int j = 0; j < 8; ++j) { b[j] = lo[j]; b[j + 8] = hi[j]; }
  return b;
}

// ---------------------------------------------------------------------------
// Kernel 1: fused QKV projection.  C[m,n] = x[m,:] . W*[n,:] + b*[n]
// One wave -> one 16x16 tile of Q, K and V (A fragment shared by 3 WMMAs).
// ---------------------------------------------------------------------------
__global__ __launch_bounds__(256) void qkv_kernel(
    const float* __restrict__ x,
    const float* __restrict__ Wq, const float* __restrict__ bq,
    const float* __restrict__ Wk, const float* __restrict__ bk,
    const float* __restrict__ Wv, const float* __restrict__ bv,
    f16* __restrict__ Qh, f16* __restrict__ Kh, f16* __restrict__ Vh)
{
  const int lane = threadIdx.x & 31;
  const int wave = threadIdx.x >> 5;
  const int gw = blockIdx.x * 8 + wave;        // 512*64 = 32768 wave tiles
  const int nTiles = DIM / 16;                 // 64
  const int m0 = (gw / nTiles) * 16;
  const int n0 = (gw % nTiles) * 16;
  const int r = lane & 15, h = lane >> 4;

  v8f cq = {}; v8f ck = {}; v8f cv = {};

  for (int k0 = 0; k0 < DIM; k0 += 32) {
    const float* ab = x + (size_t)(m0 + r) * DIM + k0 + h * 16;
    v4f a0 = *(const v4f*)(ab);
    v4f a1 = *(const v4f*)(ab + 4);
    v4f a2 = *(const v4f*)(ab + 8);
    v4f a3 = *(const v4f*)(ab + 12);
    v16h a;
#pragma unroll
    for (int j = 0; j < 4; ++j) {
      a[j]      = (f16)a0[j];
      a[j + 4]  = (f16)a1[j];
      a[j + 8]  = (f16)a2[j];
      a[j + 12] = (f16)a3[j];
    }

    auto loadBW = [&](const float* __restrict__ W) -> v16h {
      const float* bb = W + (size_t)(n0 + r) * DIM + k0 + h * 16;
      v4f b0 = *(const v4f*)(bb);
      v4f b1 = *(const v4f*)(bb + 4);
      v4f b2 = *(const v4f*)(bb + 8);
      v4f b3 = *(const v4f*)(bb + 12);
      v16h b;
#pragma unroll
      for (int j = 0; j < 4; ++j) {
        b[j]      = (f16)b0[j];
        b[j + 4]  = (f16)b1[j];
        b[j + 8]  = (f16)b2[j];
        b[j + 12] = (f16)b3[j];
      }
      return b;
    };

    cq = __builtin_amdgcn_wmma_f32_16x16x32_f16(false, a, false, loadBW(Wq), (short)0, cq, false, false);
    ck = __builtin_amdgcn_wmma_f32_16x16x32_f16(false, a, false, loadBW(Wk), (short)0, ck, false, false);
    cv = __builtin_amdgcn_wmma_f32_16x16x32_f16(false, a, false, loadBW(Wv), (short)0, cv, false, false);
  }

  const float bqv = bq[n0 + r], bkv = bk[n0 + r], bvv = bv[n0 + r];
#pragma unroll
  for (int i = 0; i < 8; ++i) {
    const size_t off = (size_t)(m0 + i + 8 * h) * DIM + n0 + r;
    Qh[off] = (f16)(cq[i] + bqv);
    Kh[off] = (f16)(ck[i] + bkv);
    Vh[off] = (f16)(cv[i] + bvv);
  }
}

// ---------------------------------------------------------------------------
// Kernel 2: S[b] = scale * Q[b] @ K[b]^T.  32x32 tile per wave (4 WMMA/step).
// ---------------------------------------------------------------------------
__global__ __launch_bounds__(256) void scores_kernel(
    const f16* __restrict__ Qh, const f16* __restrict__ Kh, f16* __restrict__ S)
{
  const int lane = threadIdx.x & 31;
  const int wave = threadIdx.x >> 5;
  const int gw = blockIdx.x * 8 + wave;        // 4 * 64 * 64 = 16384 wave tiles
  const int wavesPerBatch = 64 * 64;
  const int b  = gw / wavesPerBatch;
  const int t  = gw % wavesPerBatch;
  const int m0 = (t / 64) * 32;
  const int n0 = (t % 64) * 32;
  const f16* Qb = Qh + (size_t)b * SEQ * DIM;
  const f16* Kb = Kh + (size_t)b * SEQ * DIM;
  f16* Sb = S + (size_t)b * SEQ * SEQ;
  const int r = lane & 15, h = lane >> 4;

  v8f acc00 = {}, acc01 = {}, acc10 = {}, acc11 = {};

  for (int k0 = 0; k0 < DIM; k0 += 32) {
    v16h a0 = load_a_f16(Qb, DIM, m0,      k0, r, h);
    v16h a1 = load_a_f16(Qb, DIM, m0 + 16, k0, r, h);
    v16h b0 = load_b_rowk_f16(Kb, DIM, n0,      k0, r, h);
    v16h b1 = load_b_rowk_f16(Kb, DIM, n0 + 16, k0, r, h);
    acc00 = __builtin_amdgcn_wmma_f32_16x16x32_f16(false, a0, false, b0, (short)0, acc00, false, false);
    acc01 = __builtin_amdgcn_wmma_f32_16x16x32_f16(false, a0, false, b1, (short)0, acc01, false, false);
    acc10 = __builtin_amdgcn_wmma_f32_16x16x32_f16(false, a1, false, b0, (short)0, acc10, false, false);
    acc11 = __builtin_amdgcn_wmma_f32_16x16x32_f16(false, a1, false, b1, (short)0, acc11, false, false);
  }

  const float scale = 0.03125f;  // 1/sqrt(1024)
  v8f* accs[4] = { &acc00, &acc01, &acc10, &acc11 };
#pragma unroll
  for (int mi = 0; mi < 2; ++mi)
#pragma unroll
    for (int ni = 0; ni < 2; ++ni) {
      v8f& c = *accs[mi * 2 + ni];
#pragma unroll
      for (int i = 0; i < 8; ++i)
        Sb[(size_t)(m0 + mi * 16 + i + 8 * h) * SEQ + n0 + ni * 16 + r] = (f16)(c[i] * scale);
    }
}

// ---------------------------------------------------------------------------
// Kernel 3: row softmax in place over S (f16 storage, f32 math). 1 block/row.
// ---------------------------------------------------------------------------
__global__ __launch_bounds__(256) void softmax_kernel(f16* __restrict__ S)
{
  __shared__ float red[256];
  f16* Srow = S + (size_t)blockIdx.x * SEQ;
  const int t = threadIdx.x;

  float v[8];
  float mx = -1e30f;
#pragma unroll
  for (int i = 0; i < 8; ++i) { v[i] = (float)Srow[t + i * 256]; mx = fmaxf(mx, v[i]); }
  red[t] = mx; __syncthreads();
  for (int s = 128; s > 0; s >>= 1) { if (t < s) red[t] = fmaxf(red[t], red[t + s]); __syncthreads(); }
  mx = red[0]; __syncthreads();

  float sum = 0.f;
#pragma unroll
  for (int i = 0; i < 8; ++i) { v[i] = __expf(v[i] - mx); sum += v[i]; }
  red[t] = sum; __syncthreads();
  for (int s = 128; s > 0; s >>= 1) { if (t < s) red[t] += red[t + s]; __syncthreads(); }
  const float inv = 1.0f / red[0];
#pragma unroll
  for (int i = 0; i < 8; ++i) Srow[t + i * 256] = (f16)(v[i] * inv);
}

// ---------------------------------------------------------------------------
// Kernel 4: O[b] = P[b] @ V[b].  Block = 256 M x 32 N.
// V tiles (32 N cols x 64 K rows) are DMA'd into a double-buffered LDS tile
// by the Tensor Data Mover; tile i+1 is in flight while tile i is consumed
// (TENSORcnt is in-order: wait <=1 guarantees the current tile is resident).
// ---------------------------------------------------------------------------
__global__ __launch_bounds__(256) void av_kernel(
    const f16* __restrict__ P, const f16* __restrict__ Vh, float* __restrict__ out)
{
  __shared__ __align__(16) f16 vt[2][64 * 32];   // [buf][k][n], 8 KB
  const int lane = threadIdx.x & 31;
  const int wave = threadIdx.x >> 5;
  const int b  = blockIdx.z;
  const int n0 = blockIdx.x * 32;
  const int m0 = blockIdx.y * 256 + wave * 32;
  const f16* Pb = P  + (size_t)b * SEQ * SEQ;
  const f16* Vb = Vh + (size_t)b * SEQ * DIM;
  float* Ob = out + (size_t)b * SEQ * DIM;
  const int r = lane & 15, h = lane >> 4;

  // --- Tensor DMA descriptor group 1 (loop-invariant), CDNA5 ISA §8.4 ---
  //   workgroup_mask=0, data_size=1 (2B), no barrier/iterate/pad
  //   tensor_dim0=1024 (N), tensor_dim1=2048 (K rows)
  //   tile_dim0=32 (N), tile_dim1=64 (K rows), tensor_dim0_stride=1024
  u32x8 g1;
  g1[0] = 0x00010000u;                 // data_size = 2 bytes
  g1[1] = (unsigned)DIM << 16;         // tensor_dim0[15:0]
  g1[2] = (unsigned)SEQ << 16;         // tensor_dim0[31:16]=0 | tensor_dim1[15:0]
  g1[3] = 32u << 16;                   // tensor_dim1[31:16]=0 | tile_dim0=32
  g1[4] = 64u;                         // tile_dim1=64 | tile_dim2=0
  g1[5] = (unsigned)DIM;               // tensor_dim0_stride[31:0]
  g1[6] = 0u;                          // stride0[47:32] | stride1[15:0]
  g1[7] = 0u;
  const unsigned lds_base0 = (unsigned)(size_t)(&vt[0][0]);
  const unsigned lds_base1 = (unsigned)(size_t)(&vt[1][0]);

  auto issue_tile = [&](int k0, unsigned lds_addr) {
    // D# group 0: count=1, lds_addr, 57-bit global tile address, type=2
    const unsigned long long ga =
        (unsigned long long)(size_t)(Vb + (size_t)k0 * DIM + n0);
    u32x4 g0;
    g0[0] = 1u;
    g0[1] = lds_addr;
    g0[2] = (unsigned)ga;
    g0[3] = ((unsigned)(ga >> 32) & 0x01FFFFFFu) | 0x80000000u;  // type=2
    asm volatile("tensor_load_to_lds %0, %1" : : "s"(g0), "s"(g1) : "memory");
  };

  v8f acc00 = {}, acc01 = {}, acc10 = {}, acc11 = {};

  if (wave == 0) issue_tile(0, lds_base0);     // prologue: tile 0 in flight

  for (int k0 = 0; k0 < SEQ; k0 += 64) {
    const int buf = (k0 >> 6) & 1;
    if (wave == 0) {
      if (k0 + 64 < SEQ) {
        issue_tile(k0 + 64, buf ? lds_base0 : lds_base1);  // next tile in flight
        __builtin_amdgcn_s_wait_tensorcnt(1);              // current tile resident
      } else {
        __builtin_amdgcn_s_wait_tensorcnt(0);              // last tile resident
      }
    }
    __syncthreads();

    // hint next P block into cache while we compute
    __builtin_prefetch(&Pb[(size_t)(m0 + r) * SEQ + k0 + 64], 0, 0);

    const f16* vb = &vt[buf][0];
#pragma unroll
    for (int kk = 0; kk < 64; kk += 32) {
      v16h a0 = load_a_f16(Pb, SEQ, m0,      k0 + kk, r, h);
      v16h a1 = load_a_f16(Pb, SEQ, m0 + 16, k0 + kk, r, h);
      v16h b0, b1;
#pragma unroll
      for (int j = 0; j < 16; ++j) {
        b0[j] = vb[(kk + h * 16 + j) * 32 + r];
        b1[j] = vb[(kk + h * 16 + j) * 32 + 16 + r];
      }
      acc00 = __builtin_amdgcn_wmma_f32_16x16x32_f16(false, a0, false, b0, (short)0, acc00, false, false);
      acc01 = __builtin_amdgcn_wmma_f32_16x16x32_f16(false, a0, false, b1, (short)0, acc01, false, false);
      acc10 = __builtin_amdgcn_wmma_f32_16x16x32_f16(false, a1, false, b0, (short)0, acc10, false, false);
      acc11 = __builtin_amdgcn_wmma_f32_16x16x32_f16(false, a1, false, b1, (short)0, acc11, false, false);
    }
    __syncthreads();   // all waves done with vt[buf] before it is overwritten
  }

  v8f* accs[4] = { &acc00, &acc01, &acc10, &acc11 };
#pragma unroll
  for (int mi = 0; mi < 2; ++mi)
#pragma unroll
    for (int ni = 0; ni < 2; ++ni) {
      v8f& c = *accs[mi * 2 + ni];
#pragma unroll
      for (int i = 0; i < 8; ++i)
        Ob[(size_t)(m0 + mi * 16 + i + 8 * h) * DIM + n0 + ni * 16 + r] = c[i];
    }
}

// ---------------------------------------------------------------------------
extern "C" void kernel_launch(void* const* d_in, const int* in_sizes, int n_in,
                              void* d_out, int out_size, void* d_ws, size_t ws_size,
                              hipStream_t stream)
{
  const float* x  = (const float*)d_in[0];
  const float* Wq = (const float*)d_in[1];
  const float* bq = (const float*)d_in[2];
  const float* Wk = (const float*)d_in[3];
  const float* bk = (const float*)d_in[4];
  const float* Wv = (const float*)d_in[5];
  const float* bv = (const float*)d_in[6];
  float* out = (float*)d_out;

  char* ws = (char*)d_ws;
  f16* Qh = (f16*)(ws);                                    // 16 MB
  f16* Kh = (f16*)(ws + (size_t)16 * 1024 * 1024);         // 16 MB
  f16* Vh = (f16*)(ws + (size_t)32 * 1024 * 1024);         // 16 MB
  f16* S  = (f16*)(ws + (size_t)48 * 1024 * 1024);         // 32 MB (scores -> attn in place)

  qkv_kernel<<<4096, 256, 0, stream>>>(x, Wq, bq, Wk, bk, Wv, bv, Qh, Kh, Vh);
  scores_kernel<<<2048, 256, 0, stream>>>(Qh, Kh, S);
  softmax_kernel<<<BATCH * SEQ, 256, 0, stream>>>(S);
  av_kernel<<<dim3(DIM / 32, SEQ / 256, BATCH), 256, 0, stream>>>(S, Vh, out);
}
